// MultiHeadSelfAttention_57526791963168
// MI455X (gfx1250) — matrix operand order repaired and checked by reading, not compiled, not run on hardware
//
#include <hip/hip_runtime.h>
#include <stdint.h>

typedef __attribute__((ext_vector_type(16))) __bf16 v16bf;
typedef __attribute__((ext_vector_type(2)))  __bf16 v2bf;
typedef __attribute__((ext_vector_type(8)))  float  v8f;
typedef __attribute__((ext_vector_type(2)))  int    v2i;

typedef __attribute__((address_space(1))) v2i v2i_g;   // global b64 payload
typedef __attribute__((address_space(3))) v2i v2i_l;   // LDS b64 payload

#if __has_builtin(__builtin_amdgcn_global_load_async_to_lds_b64)
#define USE_ASYNC_LDS 1
#else
#define USE_ASYNC_LDS 0
#endif

namespace {

constexpr int kB = 128, kS = 512, kC = 64, kH = 4, kD = 32;

union AF { v16bf v; uint32_t u[8]; uint16_t h[16]; };

__device__ __forceinline__ uint16_t f2bf(float f) {
  union { __bf16 b; uint16_t u; } x;
  x.b = (__bf16)f;                       // native v_cvt to bf16 (RNE)
  return x.u;
}
__device__ __forceinline__ uint32_t pack2(float a, float b) {
#if __has_builtin(__builtin_amdgcn_cvt_pk_bf16_f32)
  union { v2bf v; uint32_t u; } x;
  x.v = __builtin_amdgcn_cvt_pk_bf16_f32(a, b);
  return x.u;
#else
  return (uint32_t)f2bf(a) | ((uint32_t)f2bf(b) << 16);
#endif
}
__device__ __forceinline__ float fast_rcp(float x) {
#if __has_builtin(__builtin_amdgcn_rcpf)
  return __builtin_amdgcn_rcpf(x);
#else
  return 1.0f / x;
#endif
}
__device__ __forceinline__ v8f v8f_splat(float x) {
  v8f r;
  #pragma unroll
  for (int i = 0; i < 8; ++i) r[i] = x;
  return r;
}
__device__ __forceinline__ v8f wmma_bf16(const AF& a, const AF& b, v8f c) {
  return __builtin_amdgcn_wmma_f32_16x16x32_bf16(false, a.v, false, b.v,
                                                 (short)0, c, false, false);
}
// A-fragment K-base for VGPR r, lane-half lh (contiguous k order, k = 16*lh + 2r)
__device__ __forceinline__ int a_kbase(int r, int lh) {
  return (lh << 4) + (r << 1);
}

} // namespace

// ---------------- Kernel 1: fused QKV projection (+bias, Q pre-scaled) -----
__global__ __launch_bounds__(256) void qkv_kernel(
    const float* __restrict__ X,
    const float* __restrict__ Wq, const float* __restrict__ bq,
    const float* __restrict__ Wk, const float* __restrict__ bk,
    const float* __restrict__ Wv, const float* __restrict__ bv,
    uint16_t* __restrict__ qo, uint16_t* __restrict__ ko,
    uint16_t* __restrict__ vo) {
  __shared__ uint16_t Wt[3][kD][68];   // transposed weights, padded rows
  __shared__ uint16_t Xs[8][16][kC];   // per-wave X tile (bf16)

  const int bh = blockIdx.x, b = bh >> 2, h = bh & 3;
  const int tid = threadIdx.x, wave = tid >> 5, lane = tid & 31;
  const int lm = lane & 15, lh = lane >> 4;

  const float* wsrc[3] = {Wq + h * kC * kD, Wk + h * kC * kD, Wv + h * kC * kD};
  #pragma unroll
  for (int p = 0; p < 3; ++p)
    for (int i = tid; i < kC * kD; i += 256) {
      int c = i >> 5, d = i & 31;
      Wt[p][d][c] = f2bf(wsrc[p][c * kD + d]);
    }
  const float* bias[3] = {bq, bk, bv};
  float bcol[3][2];
  #pragma unroll
  for (int p = 0; p < 3; ++p) {
    bcol[p][0] = bias[p][h * kD + lm];
    bcol[p][1] = bias[p][h * kD + 16 + lm];
  }
  uint16_t* dst[3] = {qo, ko, vo};
  const float qscale = 0.17677669529663687f;  // 1/sqrt(D)

  __syncthreads();

  for (int t = 0; t < 4; ++t) {
    const int s0 = wave * 64 + t * 16;
    const float* xrow = X + ((size_t)b * kS + s0) * kC;
    #pragma unroll
    for (int it = 0; it < 16; ++it) {
      int pi = lane + 32 * it;                 // pair index 0..511
      int row = pi >> 5, cp = (pi & 31) << 1;
      float2 xv = *(const float2*)(xrow + row * kC + cp);
      *(uint32_t*)&Xs[wave][row][cp] = pack2(xv.x, xv.y);
    }
    __syncthreads();

    AF a[2];
    #pragma unroll
    for (int kc = 0; kc < 2; ++kc)
      #pragma unroll
      for (int r = 0; r < 8; ++r)
        a[kc].u[r] = *(const uint32_t*)&Xs[wave][lm][kc * 32 + a_kbase(r, lh)];

    #pragma unroll
    for (int p = 0; p < 3; ++p) {
      #pragma unroll
      for (int nt = 0; nt < 2; ++nt) {
        v8f acc = v8f_splat(bcol[p][nt]);
        #pragma unroll
        for (int kc = 0; kc < 2; ++kc) {
          AF bf;
          #pragma unroll
          for (int r = 0; r < 8; ++r)
            bf.u[r] = *(const uint32_t*)&Wt[p][nt * 16 + lm]
                                          [kc * 32 + (lh << 4) + (r << 1)];
          acc = wmma_bf16(a[kc], bf, acc);
        }
        if (p == 0) {
          #pragma unroll
          for (int vv = 0; vv < 8; ++vv) acc[vv] *= qscale;
        }
        uint16_t* o = dst[p] + (size_t)bh * kS * kD;
        #pragma unroll
        for (int vv = 0; vv < 8; ++vv)
          o[(s0 + vv + 8 * lh) * kD + nt * 16 + lm] = f2bf(acc[vv]);
      }
    }
    __syncthreads();
  }
}

// ---------------- Kernel 2: fused flash attention per (b,h) ----------------
__global__ __launch_bounds__(256) void attn_kernel(
    const uint16_t* __restrict__ qg, const uint16_t* __restrict__ kg,
    const uint16_t* __restrict__ vg, uint16_t* __restrict__ cws) {
  __shared__ uint16_t Vs[kS][36];      // V staged, padded rows (no bank conflict)
  __shared__ uint16_t Ps[8][16][32];   // per-wave P scratch (C->A layout swap)

  const int bh = blockIdx.x, b = bh >> 2, h = bh & 3;
  const int tid = threadIdx.x, wave = tid >> 5, lane = tid & 31;
  const int lm = lane & 15, lh = lane >> 4;

  const uint16_t* kbp = kg + (size_t)bh * kS * kD;
  const uint16_t* vbp = vg + (size_t)bh * kS * kD;

#if USE_ASYNC_LDS
  // Async DMA global->LDS (ASYNCcnt), 8-byte chunks; rows stay 8B aligned
  // on both sides (global stride 64B, LDS stride 72B).
  for (int i = tid; i < kS * 8; i += 256) {     // 4096 chunks of 8 bytes
    int row = i >> 3, ch = (i & 7) << 2;        // ch: element offset (4 elems)
    __builtin_amdgcn_global_load_async_to_lds_b64(
        (v2i_g*)(vbp + row * kD + ch),
        (v2i_l*)&Vs[row][ch], 0, 0);
  }
#if __has_builtin(__builtin_amdgcn_s_wait_asynccnt)
  __builtin_amdgcn_s_wait_asynccnt(0);
#else
  asm volatile("s_wait_asynccnt 0x0" ::: "memory");
#endif
#else
  for (int i = tid; i < kS * kD / 2; i += 256) {
    int row = i >> 4, cp = (i & 15) << 1;
    *(uint32_t*)&Vs[row][cp] = *(const uint32_t*)&vbp[row * kD + cp];
  }
#endif
  __syncthreads();

  for (int t = 0; t < 4; ++t) {
    const int s0 = (wave + t * 8) * 16;
    AF qa;
    #pragma unroll
    for (int r = 0; r < 8; ++r)
      qa.u[r] = *(const uint32_t*)&qg[((size_t)bh * kS + s0 + lm) * kD +
                                      a_kbase(r, lh)];
    v8f acc0 = v8f_splat(0.f), acc1 = v8f_splat(0.f);
    float m[8], l[8];
    #pragma unroll
    for (int vv = 0; vv < 8; ++vv) { m[vv] = -1e30f; l[vv] = 0.f; }

    for (int kc = 0; kc < kS; kc += 32) {
      AF kf0, kf1;
      #pragma unroll
      for (int r = 0; r < 8; ++r) {
        kf0.u[r] = *(const uint32_t*)&kbp[(size_t)(kc + lm) * kD +
                                          (lh << 4) + (r << 1)];
        kf1.u[r] = *(const uint32_t*)&kbp[(size_t)(kc + 16 + lm) * kD +
                                          (lh << 4) + (r << 1)];
      }
      v8f s0v = wmma_bf16(qa, kf0, v8f_splat(0.f));
      v8f s1v = wmma_bf16(qa, kf1, v8f_splat(0.f));

      #pragma unroll
      for (int vv = 0; vv < 8; ++vv) {
        float tmax = fmaxf(s0v[vv], s1v[vv]);
        #pragma unroll
        for (int off = 1; off < 16; off <<= 1)
          tmax = fmaxf(tmax, __shfl_xor(tmax, off, 32));
        float nm = fmaxf(m[vv], tmax);
        float corr = __expf(m[vv] - nm);
        m[vv] = nm;
        float p0 = __expf(s0v[vv] - nm);
        float p1 = __expf(s1v[vv] - nm);
        float rs = p0 + p1;
        #pragma unroll
        for (int off = 1; off < 16; off <<= 1)
          rs += __shfl_xor(rs, off, 32);
        l[vv] = l[vv] * corr + rs;
        acc0[vv] *= corr;
        acc1[vv] *= corr;
        Ps[wave][vv + 8 * lh][lm] = f2bf(p0);
        Ps[wave][vv + 8 * lh][16 + lm] = f2bf(p1);
      }
      __asm__ volatile("" ::: "memory");   // order DS stores before DS loads

      AF pa;
      #pragma unroll
      for (int r = 0; r < 8; ++r)
        pa.u[r] = *(const uint32_t*)&Ps[wave][lm][a_kbase(r, lh)];
      AF vf0, vf1;
      #pragma unroll
      for (int e = 0; e < 16; ++e) {
        vf0.h[e] = Vs[kc + (lh << 4) + e][lm];
        vf1.h[e] = Vs[kc + (lh << 4) + e][16 + lm];
      }
      acc0 = wmma_bf16(pa, vf0, acc0);
      acc1 = wmma_bf16(pa, vf1, acc1);
      __asm__ volatile("" ::: "memory");
    }

    uint16_t* orow = cws + ((size_t)b * kS + s0) * (kH * kD) + h * kD;
    #pragma unroll
    for (int vv = 0; vv < 8; ++vv) {
      float inv = fast_rcp(l[vv]);
      orow[(vv + 8 * lh) * (kH * kD) + lm]      = f2bf(acc0[vv] * inv);
      orow[(vv + 8 * lh) * (kH * kD) + 16 + lm] = f2bf(acc1[vv] * inv);
    }
  }
}

// ---------------- Kernel 3: output projection [BS,128] @ [128,32] + bo -----
__global__ __launch_bounds__(256) void outproj_kernel(
    const uint16_t* __restrict__ cws, const float* __restrict__ Wo,
    const float* __restrict__ bo, float* __restrict__ out) {
  __shared__ uint16_t Wot[kD][132];    // transposed Wo, padded rows

  const int tid = threadIdx.x, wave = tid >> 5, lane = tid & 31;
  const int lm = lane & 15, lh = lane >> 4;

  for (int i = tid; i < (kH * kD) * kD; i += 256) {
    int kk = i >> 5, d = i & 31;
    Wot[d][kk] = f2bf(Wo[kk * kD + d]);
  }
  __syncthreads();
  float bc[2] = {bo[lm], bo[16 + lm]};

  for (int t = 0; t < 4; ++t) {
    const int tile = (blockIdx.x * 8 + wave) * 4 + t;
    const int row0 = tile * 16;
    AF a[4];
    #pragma unroll
    for (int kcI = 0; kcI < 4; ++kcI)
      #pragma unroll
      for (int r = 0; r < 8; ++r)
        a[kcI].u[r] = *(const uint32_t*)&cws[((size_t)row0 + lm) * 128 +
                                             kcI * 32 + a_kbase(r, lh)];
    #pragma unroll
    for (int nt = 0; nt < 2; ++nt) {
      v8f acc = v8f_splat(bc[nt]);
      #pragma unroll
      for (int kcI = 0; kcI < 4; ++kcI) {
        AF bf;
        #pragma unroll
        for (int r = 0; r < 8; ++r)
          bf.u[r] = *(const uint32_t*)&Wot[nt * 16 + lm]
                                         [kcI * 32 + (lh << 4) + (r << 1)];
        acc = wmma_bf16(a[kcI], bf, acc);
      }
      #pragma unroll
      for (int vv = 0; vv < 8; ++vv)
        out[((size_t)row0 + vv + 8 * lh) * kD + nt * 16 + lm] = acc[vv];
    }
  }
}

extern "C" void kernel_launch(void* const* d_in, const int* in_sizes, int n_in,
                              void* d_out, int out_size, void* d_ws,
                              size_t ws_size, hipStream_t stream) {
  (void)in_sizes; (void)n_in; (void)out_size; (void)ws_size;
  const float* X  = (const float*)d_in[0];
  const float* Wq = (const float*)d_in[1];
  const float* bq = (const float*)d_in[2];
  const float* Wk = (const float*)d_in[3];
  const float* bk = (const float*)d_in[4];
  const float* Wv = (const float*)d_in[5];
  const float* bv = (const float*)d_in[6];
  const float* Wo = (const float*)d_in[7];
  const float* bo = (const float*)d_in[8];

  const size_t n = (size_t)kB * kH * kS * kD;   // 8,388,608 elements per buf
  uint16_t* qws = (uint16_t*)d_ws;
  uint16_t* kws = qws + n;
  uint16_t* vws = kws + n;
  uint16_t* cws = vws + n;                      // [B*S, H*D] bf16 concat

  qkv_kernel<<<kB * kH, 256, 0, stream>>>(X, Wq, bq, Wk, bk, Wv, bv,
                                          qws, kws, vws);
  attn_kernel<<<kB * kH, 256, 0, stream>>>(qws, kws, vws, cws);
  outproj_kernel<<<128, 256, 0, stream>>>(cws, Wo, bo, (float*)d_out);
}
